// IdentityGateWrapper_34565896798967
// MI455X (gfx1250) — matrix operand order repaired and checked by reading, not compiled, not run on hardware
//
#include <hip/hip_runtime.h>

typedef __attribute__((ext_vector_type(16))) __bf16 v16bf;
typedef __attribute__((ext_vector_type(8)))  float  v8f;

#define DHID 2048
#define NEXP 64
#define LDS_STRIDE 68          // 64 + 4 floats pad: row stride % 64 banks == 4
#define WAVES_PER_BLOCK 4
#define TOKENS_PER_WAVE 16
#define TOKENS_PER_BLOCK (WAVES_PER_BLOCK * TOKENS_PER_WAVE)

union BfPack { uint4 u[2]; v16bf v; };

struct ABuf { float4 f0, f1, f2, f3; };           // 16 raw fp32 A elements
struct BBuf { BfPack t0, t1, t2, t3; };           // 4 pre-swizzled B tiles

// ---------------------------------------------------------------------------
// Kernel 0: convert fp32 weight [64, 2048] -> bf16, pre-swizzled into the
// exact 32x16 B-operand VGPR layout of v_wmma_f32_16x16x32_bf16:
//   tile (kb, et): lane L holds 16 K values in two runs of 8
//     col  = L & 15  (expert within tile),  half = L >> 4
//     elem e (0..7): B[k = kb*32 + half*8 + e], e (8..15): B[k = kb*32 + 16 + half*8 + (e-8)]; expert = et*16 + col
// Stored lane-major: 32 bytes per lane, 1 KB per tile, tile id = kb*4 + et.
// ---------------------------------------------------------------------------
__global__ __launch_bounds__(128)
void router_swizzle_weight(const float* __restrict__ w, uint4* __restrict__ wsw)
{
    const int gid  = blockIdx.x * blockDim.x + threadIdx.x;   // 0 .. 8191
    const int tile = gid >> 5;
    const int lane = gid & 31;
    const int kb   = tile >> 2;
    const int et   = tile & 3;
    const int col  = lane & 15;
    const int half = lane >> 4;

    const float4* src = (const float4*)(w + (size_t)(et * 16 + col) * DHID
                                          + kb * 32 + half * 8);
    float4 f0 = src[0], f1 = src[1], f2 = src[4], f3 = src[5];

    v16bf b;
    b[0]  = (__bf16)f0.x; b[1]  = (__bf16)f0.y; b[2]  = (__bf16)f0.z; b[3]  = (__bf16)f0.w;
    b[4]  = (__bf16)f1.x; b[5]  = (__bf16)f1.y; b[6]  = (__bf16)f1.z; b[7]  = (__bf16)f1.w;
    b[8]  = (__bf16)f2.x; b[9]  = (__bf16)f2.y; b[10] = (__bf16)f2.z; b[11] = (__bf16)f2.w;
    b[12] = (__bf16)f3.x; b[13] = (__bf16)f3.y; b[14] = (__bf16)f3.z; b[15] = (__bf16)f3.w;

    BfPack pk; pk.v = b;
    uint4* dst = wsw + (size_t)(tile * 32 + lane) * 2;
    dst[0] = pk.u[0];
    dst[1] = pk.u[1];
}

// ---------------------------------------------------------------------------
// Main kernel: per wave, GEMM 16 tokens x 64 experts over K=2048 using
// v_wmma_f32_16x16x32_bf16 (double-buffered, software-pipelined loads),
// then softmax + top-8 (2 lanes per token).
// ---------------------------------------------------------------------------
__device__ __forceinline__ void load_a(ABuf& a, const float* __restrict__ Ab, int kb)
{
    const float4* p = (const float4*)(Ab + kb * 32);
    a.f0 = p[0];  a.f1 = p[1];    // K = kb*32 + half*8 .. +7
    a.f2 = p[4];  a.f3 = p[5];    // K = kb*32 + 16 + half*8 .. +7
    // pull the A stream toward L2 ~4KB ahead (speculative, OOB-safe)
    __builtin_prefetch((const char*)p + 4096, 0, 3);
}

__device__ __forceinline__ void load_b(BBuf& b, const uint4* __restrict__ bpl, int kb)
{
    const uint4* p = bpl + (size_t)kb * 256;      // 4 tiles * 64 uint4 per kb
    b.t0.u[0] = p[0];   b.t0.u[1] = p[1];
    b.t1.u[0] = p[64];  b.t1.u[1] = p[65];
    b.t2.u[0] = p[128]; b.t2.u[1] = p[129];
    b.t3.u[0] = p[192]; b.t3.u[1] = p[193];
}

__device__ __forceinline__ v16bf cvt_a(const ABuf& r)
{
    v16bf a;
    a[0]  = (__bf16)r.f0.x; a[1]  = (__bf16)r.f0.y; a[2]  = (__bf16)r.f0.z; a[3]  = (__bf16)r.f0.w;
    a[4]  = (__bf16)r.f1.x; a[5]  = (__bf16)r.f1.y; a[6]  = (__bf16)r.f1.z; a[7]  = (__bf16)r.f1.w;
    a[8]  = (__bf16)r.f2.x; a[9]  = (__bf16)r.f2.y; a[10] = (__bf16)r.f2.z; a[11] = (__bf16)r.f2.w;
    a[12] = (__bf16)r.f3.x; a[13] = (__bf16)r.f3.y; a[14] = (__bf16)r.f3.z; a[15] = (__bf16)r.f3.w;
    return a;
}

__global__ __launch_bounds__(128)
void router_gemm_softmax_topk(const float* __restrict__ x,
                              const uint4* __restrict__ wsw,
                              float* __restrict__ probs,
                              float* __restrict__ topv,
                              float* __restrict__ topi,
                              int ntok)
{
    __shared__ float lds[WAVES_PER_BLOCK * TOKENS_PER_WAVE * LDS_STRIDE];

    const int tid  = threadIdx.x;
    const int wave = tid >> 5;
    const int lane = tid & 31;
    const int col  = lane & 15;    // A: row of tile; C: column
    const int half = lane >> 4;

    const int token_base = blockIdx.x * TOKENS_PER_BLOCK + wave * TOKENS_PER_WAVE;

    // ---------------- Phase 1: WMMA GEMM (double-buffered pipeline) --------
    // A layout (16-bit A 16x32): lane = row | half selects K-octet:
    //   elem e=0..7  : K = kb*32 + half*8 + e
    //   elem e=8..15 : K = kb*32 + 16 + half*8 + (e-8)
    const float* Ab  = x + (size_t)(token_base + col) * DHID + half * 8;
    const uint4* bpl = wsw + (size_t)lane * 2;

    v8f acc0 = {}, acc1 = {}, acc2 = {}, acc3 = {};

    auto compute = [&](const ABuf& ab, const BBuf& bb) {
        v16bf a = cvt_a(ab);
        acc0 = __builtin_amdgcn_wmma_f32_16x16x32_bf16(false, a, false, bb.t0.v, (short)0, acc0, false, false);
        acc1 = __builtin_amdgcn_wmma_f32_16x16x32_bf16(false, a, false, bb.t1.v, (short)0, acc1, false, false);
        acc2 = __builtin_amdgcn_wmma_f32_16x16x32_bf16(false, a, false, bb.t2.v, (short)0, acc2, false, false);
        acc3 = __builtin_amdgcn_wmma_f32_16x16x32_bf16(false, a, false, bb.t3.v, (short)0, acc3, false, false);
    };

    ABuf a0, a1; BBuf b0, b1;
    load_a(a0, Ab, 0);  load_b(b0, bpl, 0);

    #pragma unroll 1
    for (int kb = 0; kb < 62; kb += 2) {
        load_a(a1, Ab, kb + 1);  load_b(b1, bpl, kb + 1);   // prefetch next
        compute(a0, b0);                                    // consume current
        load_a(a0, Ab, kb + 2);  load_b(b0, bpl, kb + 2);
        compute(a1, b1);
    }
    load_a(a1, Ab, 63);  load_b(b1, bpl, 63);
    compute(a0, b0);     // kb = 62
    compute(a1, b1);     // kb = 63

    // ---------------- Phase 1b: transpose logits through LDS ----------------
    // C/D layout: VGPR r holds row (r + 8*half), column = col.
    float* wlds = lds + wave * TOKENS_PER_WAVE * LDS_STRIDE;
    #pragma unroll
    for (int r = 0; r < 8; ++r) {
        const int m = r + 8 * half;
        wlds[m * LDS_STRIDE +  0 + col] = acc0[r];
        wlds[m * LDS_STRIDE + 16 + col] = acc1[r];
        wlds[m * LDS_STRIDE + 32 + col] = acc2[r];
        wlds[m * LDS_STRIDE + 48 + col] = acc3[r];
    }
    __syncthreads();

    // ---------------- Phase 2: softmax + top-8, 2 lanes per token ----------
    const int t     = lane >> 1;           // token within wave
    const int h2    = lane & 1;            // which 32-expert half this lane scans
    const int tglob = token_base + t;

    float*  Lrow  = wlds + t * LDS_STRIDE + h2 * 32;
    float4* Lrow4 = (float4*)Lrow;         // 16B-aligned: stride 68 & offsets %4==0

    // max
    float m = -3.4e38f;
    #pragma unroll
    for (int j = 0; j < 8; ++j) {
        float4 v = Lrow4[j];
        m = fmaxf(m, fmaxf(fmaxf(v.x, v.y), fmaxf(v.z, v.w)));
    }
    m = fmaxf(m, __shfl_xor(m, 1, 32));

    // exp + sum (store exp back to LDS)
    float s = 0.f;
    #pragma unroll
    for (int j = 0; j < 8; ++j) {
        float4 v = Lrow4[j];
        v.x = __expf(v.x - m); v.y = __expf(v.y - m);
        v.z = __expf(v.z - m); v.w = __expf(v.w - m);
        Lrow4[j] = v;
        s += v.x + v.y + v.z + v.w;
    }
    s += __shfl_xor(s, 1, 32);
    const float inv = 1.0f / s;

    // normalize: write probs to global + keep in LDS for top-k
    float4* pout = (float4*)(probs + (size_t)tglob * NEXP + h2 * 32);
    #pragma unroll
    for (int j = 0; j < 8; ++j) {
        float4 v = Lrow4[j];
        v.x *= inv; v.y *= inv; v.z *= inv; v.w *= inv;
        Lrow4[j] = v;
        pout[j]  = v;
    }
    __asm__ volatile("" ::: "memory");

    // top-8: masked argmax passes; tie -> lowest index (matches jax top_k)
    for (int pass = 0; pass < 8; ++pass) {
        float bv = -1.0f; int bi = 0;
        #pragma unroll
        for (int j = 0; j < 8; ++j) {
            float4 v = Lrow4[j];
            const int base = h2 * 32 + j * 4;
            if (v.x > bv) { bv = v.x; bi = base;     }
            if (v.y > bv) { bv = v.y; bi = base + 1; }
            if (v.z > bv) { bv = v.z; bi = base + 2; }
            if (v.w > bv) { bv = v.w; bi = base + 3; }
        }
        const float ov = __shfl_xor(bv, 1, 32);
        const int   oi = __shfl_xor(bi, 1, 32);
        if (ov > bv || (ov == bv && oi < bi)) { bv = ov; bi = oi; }

        if (h2 == 0) {
            topv[(size_t)tglob * 8 + pass] = bv;
            topi[(size_t)tglob * 8 + pass] = (float)bi;
        }
        // owner lane masks the winner out (probs >= 0, so -1 excludes)
        if ((bi >> 5) == h2) Lrow[bi & 31] = -1.0f;
        __asm__ volatile("" ::: "memory");
    }
    (void)ntok;
}

extern "C" void kernel_launch(void* const* d_in, const int* in_sizes, int n_in,
                              void* d_out, int out_size, void* d_ws, size_t ws_size,
                              hipStream_t stream)
{
    const float* x = (const float*)d_in[0];   // [4*8192, 2048] fp32
    const float* w = (const float*)d_in[1];   // [64, 2048] fp32

    const int ntok = in_sizes[0] / DHID;      // 32768

    float* probs = (float*)d_out;                       // [ntok, 64]
    float* topv  = probs + (size_t)ntok * NEXP;         // [ntok, 8]
    float* topi  = topv  + (size_t)ntok * 8;            // [ntok, 8] (as float)

    uint4* wsw = (uint4*)d_ws;                // 256 KB pre-swizzled bf16 weight

    // 256 tiles * 32 lanes = 8192 threads
    router_swizzle_weight<<<8192 / 128, 128, 0, stream>>>(w, wsw);

    const int blocks = ntok / TOKENS_PER_BLOCK;   // 512
    router_gemm_softmax_topk<<<blocks, 128, 0, stream>>>(
        x, (const uint4*)wsw, probs, topv, topi, ntok);
}
